// WindowAttention_7567732375799
// MI455X (gfx1250) — compile-verified, matrix operand order checked by reading, not run on hardware
//
#include <hip/hip_runtime.h>

typedef _Float16 f16;
typedef __attribute__((ext_vector_type(16))) _Float16 v16h;
typedef __attribute__((ext_vector_type(8)))  _Float16 v8h;
typedef __attribute__((ext_vector_type(4)))  _Float16 v4h;
typedef __attribute__((ext_vector_type(8)))  float    v8f;

union V16 { v16h v; v8h h[2]; f16 e[16]; };

#define WQ_K      184                  // padded K stride for Wqkv (K<=183 stored; 184..191 overrun hits x-zeros)
#define WP_K      192                  // full K stride for Wproj (head-5 chunk reads K up to 191)
#define WQ_HALFS  (576 * WQ_K)         // 105984
#define WP_HALFS  (192 * WP_K)         // 36864
#define LDS_COPY_BYTES (WQ_HALFS*2 + WP_HALFS*2 + 576*4 + 192*4)   // 288768
#define PER_WAVE_BYTES 9216            // xw 6144 + qs 1024 + ks 1024 + vT 1024 (attn buf aliases ks)
#define NWAVES 4
#define LDS_TOTAL (LDS_COPY_BYTES + NWAVES*PER_WAVE_BYTES)         // 325632 <= 327680
#define SCALE 0.18257418583505537f     // 30^-0.5

#define WMMA_F16(Aop, Bop, Cop) \
    __builtin_amdgcn_wmma_f32_16x16x32_f16(false, (Aop), false, (Bop), (short)0, (Cop), false, false)

// ---------------- weight/bias packing kernel ----------------
// Packs (into d_ws, contiguously, exactly in the LDS image order):
//   WqT [p=576][k=184] f16 : column-major qkv weights, head-dim padded 30->32, q columns pre-scaled
//   WpP [c=192][k=192] f16 : proj weights, K remapped to padded head layout (h*32+d)
//   bqP [576] f32, bpP [192] f32, biasP [6][16][16] f32 (bias_table gathered via rel_index)
__global__ void prep_kernel(const float* __restrict__ w_qkv, const float* __restrict__ b_qkv,
                            const float* __restrict__ w_proj, const float* __restrict__ b_proj,
                            const float* __restrict__ bias_table, const int* __restrict__ rel_index,
                            f16* __restrict__ WqT, f16* __restrict__ WpP,
                            float* __restrict__ bqP, float* __restrict__ bpP,
                            float* __restrict__ biasP)
{
    int i = blockIdx.x * blockDim.x + threadIdx.x;
    if (i < WQ_HALFS) {
        int p = i / WQ_K, kk = i % WQ_K;
        int mat = p / 192, rem = p % 192, h = rem / 32, d = rem % 32;
        float v = 0.f;
        if (d < 30 && kk < 180) {
            int o = (mat * 6 + h) * 30 + d;
            v = w_qkv[kk * 540 + o];
            if (mat == 0) v *= SCALE;
        }
        WqT[i] = (f16)v;
        return;
    }
    i -= WQ_HALFS;
    if (i < WP_HALFS) {
        int c = i / WP_K, kk = i % WP_K;
        int h = kk / 32, d = kk % 32;
        float v = 0.f;
        if (d < 30 && c < 180) v = w_proj[(h * 30 + d) * 180 + c];
        WpP[i] = (f16)v;
        return;
    }
    i -= WP_HALFS;
    if (i < 576) {
        int mat = i / 192, rem = i % 192, h = rem / 32, d = rem % 32;
        float v = 0.f;
        if (d < 30) { v = b_qkv[(mat * 6 + h) * 30 + d]; if (mat == 0) v *= SCALE; }
        bqP[i] = v;
        return;
    }
    i -= 576;
    if (i < 192) { bpP[i] = (i < 180) ? b_proj[i] : 0.f; return; }
    i -= 192;
    if (i < 1536) {  // [h][n][m]
        int h = i >> 8, n = (i >> 4) & 15, m = i & 15;
        biasP[i] = bias_table[rel_index[n * 16 + m] * 6 + h];
    }
}

// ---------------- fused window-attention kernel ----------------
extern __shared__ __attribute__((aligned(16))) char lds_raw[];

__launch_bounds__(128, 1)
__global__ void swin_attn_kernel(const float* __restrict__ x,
                                 const f16* __restrict__ wsW,   // packed WqT|WpP|bq|bp image
                                 const float* __restrict__ biasP,
                                 float* __restrict__ out, int nwin)
{
    // one-time: stream the packed weight image into LDS (128 threads, 16B chunks)
    {
        const uint4* src = (const uint4*)wsW;
        uint4* dst = (uint4*)lds_raw;
        for (int i = threadIdx.x; i < LDS_COPY_BYTES / 16; i += 128) dst[i] = src[i];
    }
    __syncthreads();

    f16* Wq = (f16*)lds_raw;
    f16* Wp = Wq + WQ_HALFS;
    const float* bq = (const float*)(lds_raw + (WQ_HALFS + WP_HALFS) * 2);
    const float* bp = bq + 576;

    const int wave = threadIdx.x >> 5;
    const int lane = threadIdx.x & 31;
    const int lc = lane & 15;          // column / M-sub index
    const int g  = lane >> 4;          // lane-group (ISA 16-bit operand K-interleave)

    f16* xw = (f16*)(lds_raw + LDS_COPY_BYTES + wave * PER_WAVE_BYTES); // [16][192] f16 x
    f16* qs = xw + 16 * 192;   // [16][32] q (row-major); reused as head-out buffer
    f16* ks = qs + 512;        // [16][32] k (row-major); head reused as attn-prob [16][16]
    f16* vT = ks + 512;        // [32][16] v transposed
    f16* as_ = ks;             // attn probs alias (k dead after q.k^T WMMA)

    v8h zero8;
#pragma unroll
    for (int z = 0; z < 8; ++z) zero8[z] = (f16)0.f;

    // K-pad columns 180..191 of xw: staging never touches them, zero once per wave
    if (lane < 16) {
        f16* r = xw + lane * 192 + 180;
#pragma unroll
        for (int z = 0; z < 12; ++z) r[z] = (f16)0.f;
    }

    const int wstride = gridDim.x * NWAVES;

    for (int win = blockIdx.x * NWAVES + wave; win < nwin; win += wstride) {
        const float* xp = x + (size_t)win * 2880;

        // prefetch this wave's NEXT window (~90 cachelines) while we compute this one
        if (win + wstride < nwin) {
            const char* nx = (const char*)(x + (size_t)(win + wstride) * 2880);
#pragma unroll
            for (int pf = 0; pf < 3; ++pf)
                __builtin_prefetch(nx + (lane + pf * 32) * 128, 0, 1);
        }

        // ---- stage x[16][180] fp32 -> f16 LDS [16][192]: float4 in, packed v4h out ----
        // 4 | 180, so every aligned float4 lies inside one token row.
        {
            const float4* xp4 = (const float4*)xp;
            for (int j = lane; j < 720; j += 32) {
                float4 f = xp4[j];
                int m = j / 45;                 // 45 float4 per row
                int d = (j - m * 45) * 4;
                v4h h4;
                h4[0] = (f16)f.x; h4[1] = (f16)f.y; h4[2] = (f16)f.z; h4[3] = (f16)f.w;
                *(v4h*)(xw + m * 192 + d) = h4;
            }
        }

        // ---- A operand (x) for all 6 K-chunks, held in registers (48 VGPRs) ----
        V16 A[6];
#pragma unroll
        for (int c = 0; c < 6; ++c) {
            const f16* p0 = xw + lc * 192 + c * 32 + 8 * g;
            A[c].h[0] = *(const v8h*)p0;
            A[c].h[1] = *(const v8h*)(p0 + 16);
        }

        // ---- persistent projection accumulators: 12 N-tiles of [16x16] f32 ----
        v8f C2[12];
#pragma unroll
        for (int t = 0; t < 12; ++t)
#pragma unroll
            for (int r = 0; r < 8; ++r) C2[t][r] = 0.f;

        for (int h = 0; h < 6; ++h) {
            // ===== GEMM1: q,k,v tile PAIRS, B prefetched 2 k-chunks ahead =====
#pragma unroll
            for (int mat = 0; mat < 3; ++mat) {
                const int colbase = mat * 192 + h * 32 + lc;
                const f16* p0 = Wq + colbase * WQ_K + 8 * g;     // d-tile 0
                const f16* p1 = p0 + 16 * WQ_K;                  // d-tile 1
                const float badd0 = bq[colbase];
                const float badd1 = bq[colbase + 16];
                V16 B0[3], B1[3];
                B0[0].h[0] = *(const v8h*)(p0);      B0[0].h[1] = *(const v8h*)(p0 + 16);
                B1[0].h[0] = *(const v8h*)(p1);      B1[0].h[1] = *(const v8h*)(p1 + 16);
                B0[1].h[0] = *(const v8h*)(p0 + 32); B0[1].h[1] = *(const v8h*)(p0 + 48);
                B1[1].h[0] = *(const v8h*)(p1 + 32); B1[1].h[1] = *(const v8h*)(p1 + 48);
                v8f C0, C1;
#pragma unroll
                for (int r = 0; r < 8; ++r) { C0[r] = 0.f; C1[r] = 0.f; }
#pragma unroll
                for (int c = 0; c < 6; ++c) {
                    if (c < 4) {   // fetch chunk c+2 into the free rotation slot
                        const f16* q0 = p0 + (c + 2) * 32;
                        const f16* q1 = p1 + (c + 2) * 32;
                        B0[(c + 2) % 3].h[0] = *(const v8h*)q0;
                        B0[(c + 2) % 3].h[1] = *(const v8h*)(q0 + 16);
                        B1[(c + 2) % 3].h[0] = *(const v8h*)q1;
                        B1[(c + 2) % 3].h[1] = *(const v8h*)(q1 + 16);
                    }
                    C0 = WMMA_F16(A[c].v, B0[c % 3].v, C0);
                    C1 = WMMA_F16(A[c].v, B1[c % 3].v, C1);
                }
                if (mat < 2) {               // q,k: row-major [token][dim]
                    f16* dst = mat ? ks : qs;
#pragma unroll
                    for (int r = 0; r < 8; ++r) {
                        dst[(r + 8 * g) * 32 + lc]      = (f16)(C0[r] + badd0);
                        dst[(r + 8 * g) * 32 + 16 + lc] = (f16)(C1[r] + badd1);
                    }
                } else {                     // v: transposed [dim][token], packed 16B stores
                    v8h P0, P1;
#pragma unroll
                    for (int r = 0; r < 8; ++r) {
                        P0[r] = (f16)(C0[r] + badd0);
                        P1[r] = (f16)(C1[r] + badd1);
                    }
                    *(v8h*)(vT + lc * 16 + 8 * g)        = P0;
                    *(v8h*)(vT + (16 + lc) * 16 + 8 * g) = P1;
                }
            }

            // ===== attention: S = q . k^T =====
            V16 qa, kb;
            { const f16* p = qs + lc * 32 + 8 * g; qa.h[0] = *(const v8h*)p; qa.h[1] = *(const v8h*)(p + 16); }
            { const f16* p = ks + lc * 32 + 8 * g; kb.h[0] = *(const v8h*)p; kb.h[1] = *(const v8h*)(p + 16); }
            v8f S;
#pragma unroll
            for (int r = 0; r < 8; ++r) S[r] = 0.f;
            S = WMMA_F16(qa.v, kb.v, S);

            // bias + softmax along key dim (16 lanes per half-wave hold one row)
            const float* bb = biasP + h * 256;
#pragma unroll
            for (int r = 0; r < 8; ++r) {
                float vv = S[r] + bb[(r + 8 * g) * 16 + lc];
                float mx = vv;
                mx = fmaxf(mx, __shfl_xor(mx, 1, 16));
                mx = fmaxf(mx, __shfl_xor(mx, 2, 16));
                mx = fmaxf(mx, __shfl_xor(mx, 4, 16));
                mx = fmaxf(mx, __shfl_xor(mx, 8, 16));
                float e = __expf(vv - mx);
                float s = e;
                s += __shfl_xor(s, 1, 16);
                s += __shfl_xor(s, 2, 16);
                s += __shfl_xor(s, 4, 16);
                s += __shfl_xor(s, 8, 16);
                float pr = __fdividef(e, s);
                as_[(r + 8 * g) * 16 + lc] = (f16)pr;
            }

            // ===== head-out = P . v  (K=16 keys zero-padded to 32 via register zeros) =====
            V16 aa;
            aa.h[0] = *(const v8h*)(as_ + lc * 16 + 8 * g);
            aa.h[1] = zero8;
            f16* hos = qs;                       // q buffer dead; reuse for head-out
            {
                V16 vb0, vb1;
                vb0.h[0] = *(const v8h*)(vT + lc * 16 + 8 * g);        vb0.h[1] = zero8;
                vb1.h[0] = *(const v8h*)(vT + (16 + lc) * 16 + 8 * g); vb1.h[1] = zero8;
                v8f H0, H1;
#pragma unroll
                for (int r = 0; r < 8; ++r) { H0[r] = 0.f; H1[r] = 0.f; }
                H0 = WMMA_F16(aa.v, vb0.v, H0);
                H1 = WMMA_F16(aa.v, vb1.v, H1);
#pragma unroll
                for (int r = 0; r < 8; ++r) {
                    hos[(r + 8 * g) * 32 + lc]      = (f16)H0[r];
                    hos[(r + 8 * g) * 32 + 16 + lc] = (f16)H1[r];
                }
            }

            // ===== projection partial: head h = K-chunk h of Wp; B prefetched 2 tiles ahead =====
            V16 ha;
            { const f16* p = hos + lc * 32 + 8 * g; ha.h[0] = *(const v8h*)p; ha.h[1] = *(const v8h*)(p + 16); }
            {
                const f16* wp0 = Wp + lc * WP_K + h * 32 + 8 * g;
                V16 Bp[3];
                Bp[0].h[0] = *(const v8h*)(wp0);
                Bp[0].h[1] = *(const v8h*)(wp0 + 16);
                Bp[1].h[0] = *(const v8h*)(wp0 + 16 * WP_K);
                Bp[1].h[1] = *(const v8h*)(wp0 + 16 * WP_K + 16);
#pragma unroll
                for (int t = 0; t < 12; ++t) {
                    if (t < 10) {
                        const f16* p = wp0 + (t + 2) * 16 * WP_K;
                        Bp[(t + 2) % 3].h[0] = *(const v8h*)p;
                        Bp[(t + 2) % 3].h[1] = *(const v8h*)(p + 16);
                    }
                    C2[t] = WMMA_F16(ha.v, Bp[t % 3].v, C2[t]);
                }
            }
        }

        // ---- epilogue: + b_proj, fp32 store (padded columns masked) ----
        float* op = out + (size_t)win * 2880;
#pragma unroll
        for (int t = 0; t < 12; ++t) {
            int c = t * 16 + lc;
            if (c < 180) {
                float b = bp[c];
#pragma unroll
                for (int r = 0; r < 8; ++r)
                    op[(r + 8 * g) * 180 + c] = C2[t][r] + b;
            }
        }
    }
}

extern "C" void kernel_launch(void* const* d_in, const int* in_sizes, int n_in,
                              void* d_out, int out_size, void* d_ws, size_t ws_size,
                              hipStream_t stream)
{
    const float* x          = (const float*)d_in[0];
    const float* w_qkv      = (const float*)d_in[1];
    const float* b_qkv      = (const float*)d_in[2];
    const float* w_proj     = (const float*)d_in[3];
    const float* b_proj     = (const float*)d_in[4];
    const float* bias_table = (const float*)d_in[5];
    const int*   rel_index  = (const int*)d_in[6];
    int nwin = in_sizes[0] / 2880;

    char* ws = (char*)d_ws;
    f16*   WqT   = (f16*)ws;
    f16*   WpP   = (f16*)(ws + WQ_HALFS * 2);
    float* bqP   = (float*)(ws + (WQ_HALFS + WP_HALFS) * 2);
    float* bpP   = bqP + 576;
    float* biasP = bpP + 192;

    const int prep_elems = WQ_HALFS + WP_HALFS + 576 + 192 + 1536;
    prep_kernel<<<(prep_elems + 255) / 256, 256, 0, stream>>>(
        w_qkv, b_qkv, w_proj, b_proj, bias_table, rel_index,
        WqT, WpP, bqP, bpP, biasP);

    hipFuncSetAttribute((const void*)swin_attn_kernel,
                        hipFuncAttributeMaxDynamicSharedMemorySize, LDS_TOTAL);
    swin_attn_kernel<<<1024, 128, LDS_TOTAL, stream>>>(x, WqT, biasP, (float*)d_out, nwin);
}
